// Adjust_VMamba_62577673503885
// MI455X (gfx1250) — matrix operand order repaired and checked by reading, not compiled, not run on hardware
//
#include <hip/hip_runtime.h>
#include <hip/hip_bf16.h>
#include <math.h>

typedef __attribute__((ext_vector_type(16))) _Float16 v16h;
typedef __attribute__((ext_vector_type(8)))  _Float16 v8h;
typedef __attribute__((ext_vector_type(8)))  float    v8f;

#define DEV __device__ __forceinline__
#define AS1 __attribute__((address_space(1)))
#define AS3 __attribute__((address_space(3)))

#if defined(__HIP_DEVICE_COMPILE__) && \
    __has_builtin(__builtin_amdgcn_global_load_async_to_lds_b32) && \
    __has_builtin(__builtin_amdgcn_s_wait_asynccnt)
#define SCAN_ASYNC 1
#else
#define SCAN_ASYNC 0
#endif

DEV float silu_f(float x) { return x / (1.0f + __expf(-x)); }
DEV float softplus_f(float x) { return (x > 20.0f) ? x : log1pf(__expf(x)); }

// ---------------------------------------------------------------- f32 -> f16
__global__ void cvt_f16_kernel(const float* __restrict__ s, _Float16* __restrict__ d, int n) {
    int i = blockIdx.x * blockDim.x + threadIdx.x;
    if (i < n) d[i] = (_Float16)s[i];
}

// -------------------------------------- f32 -> f16 with transpose (rows x cols -> cols x rows)
__global__ void cvt_f16_T_kernel(const float* __restrict__ s, _Float16* __restrict__ d,
                                 int rows, int cols) {
    int i = blockIdx.x * blockDim.x + threadIdx.x;
    if (i < rows * cols) {
        int r = i / cols, c = i % cols;
        d[(long)c * rows + r] = (_Float16)s[i];
    }
}

// ------------------------------------------------- LayerNorm over C=256, f16 out
// x is NCHW (4,256,32,32); output row-major (4096 rows x 256), row = b*1024 + h*32 + w
__global__ void ln_in_kernel(const float* __restrict__ x, const float* __restrict__ g,
                             const float* __restrict__ b, _Float16* __restrict__ out) {
    const int row = blockIdx.x;            // 4096
    const int c = threadIdx.x;             // 256
    const int bb = row >> 10, l = row & 1023;
    const int h = l >> 5, w = l & 31;
    float v = x[((bb * 256 + c) * 32 + h) * 32 + w];
    __shared__ float red[256];
    red[c] = v; __syncthreads();
    for (int s = 128; s > 0; s >>= 1) { if (c < s) red[c] += red[c + s]; __syncthreads(); }
    float mean = red[0] * (1.0f / 256.0f);
    __syncthreads();
    float d0 = v - mean;
    red[c] = d0 * d0; __syncthreads();
    for (int s = 128; s > 0; s >>= 1) { if (c < s) red[c] += red[c + s]; __syncthreads(); }
    float inv = rsqrtf(red[0] * (1.0f / 256.0f) + 1e-5f);
    out[row * 256 + c] = (_Float16)(d0 * inv * g[c] + b[c]);
}

// --------------------------------------------- generic batched WMMA f16 GEMM
// C[M,N] (f32) = A[M,K] (f16 row-major, lda=K-stride) * Bt[N,K] (f16, B stored transposed,
// ldb = K-stride). All per-lane fragment loads are contiguous 16-byte v8h loads.
__global__ void gemm_f16_wmma_kernel(const _Float16* __restrict__ A,
                                     const _Float16* __restrict__ Bt,
                                     float* __restrict__ C,
                                     int M, int N, int K, int lda, int ldb, int ldc,
                                     long sA, long sB, long sC, int aMod) {
    const int batch = blockIdx.y;
    A  += (long)(batch % aMod) * sA;
    Bt += (long)batch * sB;
    C  += (long)batch * sC;
    const int wave = (int)((blockIdx.x * blockDim.x + threadIdx.x) >> 5);
    const int lane = threadIdx.x & 31;
    const int tilesN = N >> 4, tilesM = M >> 4;
    if (wave >= tilesM * tilesN) return;           // wave-uniform, EXEC stays full
    const int m0 = (wave / tilesN) << 4;
    const int n0 = (wave % tilesN) << 4;
    const int arow = m0 + (lane & 15);
    const int bcol = n0 + (lane & 15);
    const int hi = lane >> 4;                      // lane half-select
    const _Float16* arowp = A + (long)arow * lda;
    const _Float16* bcolp = Bt + (long)bcol * ldb;
    v8f acc = {};
    #pragma unroll 4
    for (int k0 = 0; k0 < K; k0 += 32) {
        // A fragment: lane holds row=arow, K = {ka..ka+7} U {ka+16..ka+23}
        const int ka = k0 + (hi << 3);
        v8h alo = *(const v8h*)(arowp + ka);
        v8h ahi = *(const v8h*)(arowp + ka + 16);
        // B fragment: lane holds col=bcol, K = {kb..kb+7} U {kb+16..kb+23}
        const int kb = k0 + (hi << 3);
        v8h blo = *(const v8h*)(bcolp + kb);
        v8h bhi = *(const v8h*)(bcolp + kb + 16);
        v16h af, bf;
        #pragma unroll
        for (int i = 0; i < 8; i++) {
            af[i] = alo[i]; af[i + 8] = ahi[i];
            bf[i] = blo[i]; bf[i + 8] = bhi[i];
        }
        acc = __builtin_amdgcn_wmma_f32_16x16x32_f16(false, af, false, bf,
                                                     (short)0, acc, false, false);
    }
    #pragma unroll
    for (int r = 0; r < 8; r++) {
        int row = m0 + r + (hi << 3);
        C[(long)row * ldc + bcol] = acc[r];
    }
}

// --------------------- depthwise 3x3 conv + SiLU; emits all 4 scan directions
// d-fastest mapping: 9-tap reads of xz rows are coalesced across lanes.
// xs f32: (bk, d, l) for the scan; xshT f16: (bk, l, d) transposed for WMMA B operand.
__global__ void conv_dw_kernel(const float* __restrict__ xz,
                               const float* __restrict__ wc,
                               const float* __restrict__ bc,
                               float* __restrict__ xs,
                               _Float16* __restrict__ xshT) {
    int idx = blockIdx.x * blockDim.x + threadIdx.x;   // ((b*1024 + l)*512 + d)
    if (idx >= 4 * 1024 * 512) return;
    int d = idx & 511;
    int l = (idx >> 9) & 1023;
    int b = idx >> 19;
    int h = l >> 5, w = l & 31;
    float acc = 0.0f;
    #pragma unroll
    for (int ky = 0; ky < 3; ky++) {
        int hh = h + ky - 1;
        if ((unsigned)hh >= 32u) continue;
        #pragma unroll
        for (int kx = 0; kx < 3; kx++) {
            int ww = w + kx - 1;
            if ((unsigned)ww >= 32u) continue;
            acc += xz[(long)((b << 10) + (hh << 5) + ww) * 1024 + d] * wc[d * 9 + ky * 3 + kx];
        }
    }
    float v = silu_f(acc + bc[d]);
    int l1 = (w << 5) + h;
    // f32 scan layout: row (bk*512 + d) of length 1024
    long base = ((long)(b * 4) * 512 + d) << 10;
    const long stride = 512L << 10;
    xs[base + l]                        = v;
    xs[base + stride + l1]              = v;
    xs[base + 2 * stride + (1023 - l)]  = v;
    xs[base + 3 * stride + (1023 - l1)] = v;
    // f16 transposed layout for GEMM: (bk, l', d), K-dim d contiguous
    _Float16 vh = (_Float16)v;
    long tb = ((long)(b * 4) * 1024) * 512 + d;
    const long kstride = 1024L * 512;
    xshT[tb + (long)l * 512]                          = vh;
    xshT[tb + kstride + (long)l1 * 512]               = vh;
    xshT[tb + 2 * kstride + (long)(1023 - l) * 512]   = vh;
    xshT[tb + 3 * kstride + (long)(1023 - l1) * 512]  = vh;
}

// --------------------------------------- dt = softplus(dts @ w_dt^T + b_dt)
__global__ void dt_kernel(const float* __restrict__ xdbl,
                          const float* __restrict__ wdt,
                          const float* __restrict__ bdt,
                          float* __restrict__ dt) {
    long idx = (long)blockIdx.x * blockDim.x + threadIdx.x; // bk*512*1024 + d*1024 + l
    if (idx >= 16L * 512 * 1024) return;
    int l = (int)(idx & 1023);
    int d = (int)((idx >> 10) & 511);
    int bk = (int)(idx >> 19);
    int k = bk & 3;
    float acc = bdt[k * 512 + d];
    const float* xr = xdbl + ((long)bk * 48) * 1024 + l;
    const float* wr = wdt + (long)(k * 512 + d) * 16;
    #pragma unroll
    for (int r = 0; r < 16; r++) acc += xr[(long)r * 1024] * wr[r];
    dt[idx] = softplus_f(acc);
}

// ----------------- selective scan: thread = one (b,k,d) channel, 16 states
// B/C step vectors staged in LDS per 64-step tile (shared by the block's 256 d),
// using CDNA5 async global->LDS loads (ASYNCcnt-tracked, no VGPR round-trip).
__global__ void scan_kernel(const float* __restrict__ xs,
                            const float* __restrict__ dt,
                            const float* __restrict__ xdbl,
                            const float* __restrict__ Alog,
                            const float* __restrict__ Ds,
                            float* __restrict__ ys) {
    const int bk = blockIdx.y;                     // 0..15
    const int d  = blockIdx.x * 256 + threadIdx.x; // 0..511
    const int k  = bk & 3;
    float A[16];
    #pragma unroll
    for (int n = 0; n < 16; n++) A[n] = -__expf(Alog[(long)(k * 512 + d) * 16 + n]);
    const float Dd = Ds[k * 512 + d];
    float h[16];
    #pragma unroll
    for (int n = 0; n < 16; n++) h[n] = 0.0f;
    const long rowOff = ((long)bk * 512 + d) << 10;
    const float* xrow = xs + rowOff;
    const float* drow = dt + rowOff;
    float* yrow = ys + rowOff;
    const float* Bbase = xdbl + ((long)bk * 48 + 16) * 1024;
    const float* Cbase = xdbl + ((long)bk * 48 + 32) * 1024;
    __shared__ float sB[64][16];
    __shared__ float sC[64][16];
    for (int l0 = 0; l0 < 1024; l0 += 64) {
        __syncthreads();
#if SCAN_ASYNC
        for (int j = threadIdx.x; j < 1024; j += 256) {
            int t = j >> 4, n = j & 15;
            __builtin_amdgcn_global_load_async_to_lds_b32(
                (AS1 int*)(unsigned long long)(Bbase + (long)n * 1024 + l0 + t),
                (AS3 int*)(unsigned)(unsigned long long)(&sB[t][n]), 0, 0);
            __builtin_amdgcn_global_load_async_to_lds_b32(
                (AS1 int*)(unsigned long long)(Cbase + (long)n * 1024 + l0 + t),
                (AS3 int*)(unsigned)(unsigned long long)(&sC[t][n]), 0, 0);
        }
        __builtin_amdgcn_s_wait_asynccnt(0);
#else
        for (int j = threadIdx.x; j < 1024; j += 256) {
            int t = j >> 4, n = j & 15;
            sB[t][n] = Bbase[(long)n * 1024 + l0 + t];
            sC[t][n] = Cbase[(long)n * 1024 + l0 + t];
        }
#endif
        __syncthreads();
        for (int t = 0; t < 64; t++) {
            int l = l0 + t;
            float xl = xrow[l];
            float dtl = drow[l];
            float bx = dtl * xl;
            float y = 0.0f;
            #pragma unroll
            for (int n = 0; n < 16; n++) {
                h[n] = __expf(dtl * A[n]) * h[n] + bx * sB[t][n];
                y += h[n] * sC[t][n];
            }
            yrow[l] = y + Dd * xl;
        }
    }
}

// -------------- combine 4 directions + LayerNorm(Di=512) + SiLU(z) gate -> f16
__global__ void combine_ln_gate_kernel(const float* __restrict__ ys,
                                       const float* __restrict__ xz,
                                       const float* __restrict__ og,
                                       const float* __restrict__ ob,
                                       _Float16* __restrict__ yg) {
    const int row = blockIdx.x;            // 4096
    const int b = row >> 10, l = row & 1023;
    const int h = l >> 5, w = l & 31;
    const int l1 = (w << 5) + h;
    const int tid = threadIdx.x;           // 256, each owns 2 d's
    const long stride = 512L << 10;
    float v[2];
    float sum = 0.0f;
    __shared__ float red[256];
    #pragma unroll
    for (int j = 0; j < 2; j++) {
        int d = tid + j * 256;
        long base = ((long)(b * 4) * 512 + d) << 10;
        float y = ys[base + l] + ys[base + stride + l1]
                + ys[base + 2 * stride + (1023 - l)]
                + ys[base + 3 * stride + (1023 - l1)];
        v[j] = y; sum += y;
    }
    red[tid] = sum; __syncthreads();
    for (int s = 128; s > 0; s >>= 1) { if (tid < s) red[tid] += red[tid + s]; __syncthreads(); }
    float mean = red[0] * (1.0f / 512.0f);
    __syncthreads();
    float vs = 0.0f;
    #pragma unroll
    for (int j = 0; j < 2; j++) { float dd = v[j] - mean; vs += dd * dd; }
    red[tid] = vs; __syncthreads();
    for (int s = 128; s > 0; s >>= 1) { if (tid < s) red[tid] += red[tid + s]; __syncthreads(); }
    float inv = rsqrtf(red[0] * (1.0f / 512.0f) + 1e-5f);
    #pragma unroll
    for (int j = 0; j < 2; j++) {
        int d = tid + j * 256;
        float z = xz[(long)row * 1024 + 512 + d];
        float o = (v[j] - mean) * inv * og[d] + ob[d];
        yg[(long)row * 512 + d] = (_Float16)(o * silu_f(z));
    }
}

// ------- out-proj WMMA (4096x256 = yg[4096x512] @ w_out[512x256]) + residual,
// B pre-transposed (woT: 256x512); scatter-store into NCHW output layout.
__global__ void gemm_out_wmma_kernel(const _Float16* __restrict__ A,
                                     const _Float16* __restrict__ Bt,
                                     const float* __restrict__ x,
                                     float* __restrict__ out) {
    const int wave = (int)((blockIdx.x * blockDim.x + threadIdx.x) >> 5);
    const int lane = threadIdx.x & 31;
    if (wave >= 256 * 16) return;
    const int m0 = (wave >> 4) << 4;
    const int n0 = (wave & 15) << 4;
    const int arow = m0 + (lane & 15);
    const int bcol = n0 + (lane & 15);
    const int hi = lane >> 4;
    const _Float16* arowp = A + (long)arow * 512;
    const _Float16* bcolp = Bt + (long)bcol * 512;
    v8f acc = {};
    #pragma unroll 4
    for (int k0 = 0; k0 < 512; k0 += 32) {
        int ka = k0 + (hi << 3);
        v8h alo = *(const v8h*)(arowp + ka);
        v8h ahi = *(const v8h*)(arowp + ka + 16);
        int kb = k0 + (hi << 3);
        v8h blo = *(const v8h*)(bcolp + kb);
        v8h bhi = *(const v8h*)(bcolp + kb + 16);
        v16h af, bf;
        #pragma unroll
        for (int i = 0; i < 8; i++) {
            af[i] = alo[i]; af[i + 8] = ahi[i];
            bf[i] = blo[i]; bf[i + 8] = bhi[i];
        }
        acc = __builtin_amdgcn_wmma_f32_16x16x32_f16(false, af, false, bf,
                                                     (short)0, acc, false, false);
    }
    #pragma unroll
    for (int r = 0; r < 8; r++) {
        int m = m0 + r + (hi << 3);
        int b = m >> 10, l = m & 1023;
        int h = l >> 5, w = l & 31;
        long oi = (((long)(b * 256) + bcol) * 32 + h) * 32 + w;
        out[oi] = x[oi] + acc[r];
    }
}

// ---------------------------------------------------------------------------
extern "C" void kernel_launch(void* const* d_in, const int* in_sizes, int n_in,
                              void* d_out, int out_size, void* d_ws, size_t ws_size,
                              hipStream_t stream) {
    const float* x      = (const float*)d_in[0];
    const float* ln_g   = (const float*)d_in[1];
    const float* ln_b   = (const float*)d_in[2];
    const float* w_in   = (const float*)d_in[3];
    const float* w_conv = (const float*)d_in[4];
    const float* b_conv = (const float*)d_in[5];
    const float* w_xprj = (const float*)d_in[6];
    const float* w_dt   = (const float*)d_in[7];
    const float* b_dt   = (const float*)d_in[8];
    const float* A_log  = (const float*)d_in[9];
    const float* Ds     = (const float*)d_in[10];
    const float* onrm_g = (const float*)d_in[11];
    const float* onrm_b = (const float*)d_in[12];
    const float* w_out  = (const float*)d_in[13];
    float* out = (float*)d_out;

    // workspace carve-up (256B aligned)
    char* ws = (char*)d_ws;
    size_t off = 0;
    auto carve = [&](size_t bytes) -> char* {
        char* p = ws + off;
        off = (off + bytes + 255) & ~(size_t)255;
        return p;
    };
    _Float16* xnh  = (_Float16*)carve(4096UL * 256 * 2);      // LN'd input, f16
    _Float16* winT = (_Float16*)carve(1024UL * 256 * 2);      // w_in f16, transposed (N x K)
    float*    xz   = (float*)   carve(4096UL * 1024 * 4);     // in-proj out (xin|z)
    float*    xs   = (float*)   carve(16UL * 512 * 1024 * 4); // 4-dir sequences f32 (bk,d,l)
    _Float16* xshT = (_Float16*)carve(16UL * 1024 * 512 * 2); // f16 transposed (bk,l,d)
    _Float16* wxh  = (_Float16*)carve(4UL * 48 * 512 * 2);    // w_xproj f16 (A operand, row-major)
    float*    xdbl = (float*)   carve(16UL * 48 * 1024 * 4);  // x-proj out
    float*    dtb  = (float*)   carve(16UL * 512 * 1024 * 4); // softplus(dt)
    float*    ysb  = (float*)   carve(16UL * 512 * 1024 * 4); // scan out
    _Float16* ygh  = (_Float16*)carve(4096UL * 512 * 2);      // gated/normed, f16
    _Float16* woT  = (_Float16*)carve(256UL * 512 * 2);       // w_out f16, transposed (N x K)

    // 1) weight conversions (+ transpose for B operands)
    cvt_f16_T_kernel<<<(256 * 1024 + 255) / 256, 256, 0, stream>>>(w_in, winT, 256, 1024);
    cvt_f16_kernel<<<(4 * 48 * 512 + 255) / 256, 256, 0, stream>>>(w_xprj, wxh, 4 * 48 * 512);
    cvt_f16_T_kernel<<<(512 * 256 + 255) / 256, 256, 0, stream>>>(w_out, woT, 512, 256);

    // 2) LayerNorm over C
    ln_in_kernel<<<4096, 256, 0, stream>>>(x, ln_g, ln_b, xnh);

    // 3) in-proj GEMM: (4096 x 1024) = xn(4096x256) @ w_in(256x1024); B transposed, ldb=256
    gemm_f16_wmma_kernel<<<dim3(2048, 1), 256, 0, stream>>>(
        xnh, winT, xz, 4096, 1024, 256, 256, 256, 1024, 0L, 0L, 0L, 1);

    // 4) depthwise conv + SiLU + build 4 directional sequences (f32 + f16-transposed)
    conv_dw_kernel<<<(4 * 512 * 1024) / 256, 256, 0, stream>>>(xz, w_conv, b_conv, xs, xshT);

    // 5) x-proj GEMM, 16 batches (b,k): (48 x 1024) = w_xproj[k](48x512) @ xs[b,k](512x1024)
    //    B = xshT per-batch (1024 x 512), ldb=512
    gemm_f16_wmma_kernel<<<dim3(24, 16), 256, 0, stream>>>(
        wxh, xshT, xdbl, 48, 1024, 512, 512, 512, 1024,
        48L * 512, 1024L * 512, 48L * 1024, 4);

    // 6) dt projection + softplus
    dt_kernel<<<(int)((16L * 512 * 1024) / 256), 256, 0, stream>>>(xdbl, w_dt, b_dt, dtb);

    // 7) selective scan (8192 channels, async-LDS-staged B/C)
    scan_kernel<<<dim3(2, 16), 256, 0, stream>>>(xs, dtb, xdbl, A_log, Ds, ysb);

    // 8) combine directions + LayerNorm(Di) + SiLU(z) gate
    combine_ln_gate_kernel<<<4096, 256, 0, stream>>>(ysb, xz, onrm_g, onrm_b, ygh);

    // 9) out-proj GEMM + residual, NCHW scatter: 4096 tiles -> 512 blocks
    gemm_out_wmma_kernel<<<512, 256, 0, stream>>>(ygh, woT, x, out);
}
